// VariationalEncoder_57758720196620
// MI455X (gfx1250) — compile-verified, matrix operand order checked by reading, not run on hardware
//
#include <hip/hip_runtime.h>

typedef __attribute__((ext_vector_type(16))) _Float16 v16h;
typedef __attribute__((ext_vector_type(8)))  _Float16 v8h;
typedef __attribute__((ext_vector_type(8)))  float    v8f;

#define DIN 128   // input feature dim == hidden dim == GEMM K
#define DOUT 64
#define LDSP (DIN + 8)   // padded LDS row stride (136 halves = 272B, 16B-aligned)

// ---------------------------------------------------------------- utilities
__global__ void fill_kernel(float* __restrict__ p, long long n, float v) {
    long long i = (long long)blockIdx.x * blockDim.x + threadIdx.x;
    long long stride = (long long)gridDim.x * blockDim.x;
    for (; i < n; i += stride) p[i] = v;
}

__global__ void degree_kernel(const int* __restrict__ dst, int E,
                              float* __restrict__ deg) {
    int i = blockIdx.x * blockDim.x + threadIdx.x;
    if (i < E) {
        __hip_atomic_fetch_add(&deg[dst[i]], 1.0f,
                               __ATOMIC_RELAXED, __HIP_MEMORY_SCOPE_AGENT);
    }
}

__global__ void rsqrt_kernel(float* __restrict__ deg, int n) {
    int i = blockIdx.x * blockDim.x + threadIdx.x;
    // self-loops guarantee deg >= 1, matching reference's deg>0 branch
    if (i < n) deg[i] = rsqrtf(deg[i]);
}

// ------------------------------------------ weight transpose + f16 convert
// Wt1[col][k] = W1[k][col]                 (128x128)
// Wt2[col][k] = col<64 ? Wmu[k][col] : Wls[k][col-64]   (128x128)
__global__ void prep_w_kernel(const float* __restrict__ W1,
                              const float* __restrict__ Wmu,
                              const float* __restrict__ Wls,
                              _Float16* __restrict__ Wt1,
                              _Float16* __restrict__ Wt2) {
    int i = blockIdx.x * blockDim.x + threadIdx.x;
    if (i < DIN * DIN) {
        int col = i >> 7, k = i & (DIN - 1);
        Wt1[(long long)col * DIN + k] = (_Float16)W1[(long long)k * DIN + col];
        float v = (col < DOUT) ? Wmu[(long long)k * DOUT + col]
                               : Wls[(long long)k * DOUT + (col - DOUT)];
        Wt2[(long long)col * DIN + k] = (_Float16)v;
    }
}

// -------------------------------------------------- edge scatter-aggregate
// agg[dst] += feat[src] * dinv[src]*dinv[dst], 128 floats per edge.
// One wave32 per edge, float4 per lane (512B gather per edge).
// Virtual edges e in [E, E+N) are the self-loops (src == dst == e-E).
__global__ void scatter128_kernel(const float* __restrict__ feat,
                                  float* __restrict__ agg,
                                  const int* __restrict__ src,
                                  const int* __restrict__ dst,
                                  const float* __restrict__ dinv,
                                  int E, int n_nodes) {
    long long gid   = (long long)blockIdx.x * blockDim.x + threadIdx.x;
    long long nwarp = ((long long)gridDim.x * blockDim.x) >> 5;
    int lane = threadIdx.x & 31;
    long long total = (long long)E + n_nodes;
    for (long long e = gid >> 5; e < total; e += nwarp) {
        int s, d;
        if (e < (long long)E) { s = src[e]; d = dst[e]; }
        else                  { s = d = (int)(e - E); }
        float nrm = dinv[s] * dinv[d];
        float4 v = ((const float4*)(feat + (long long)s * DIN))[lane];
        float* a = agg + (long long)d * DIN + lane * 4;
        __hip_atomic_fetch_add(a + 0, v.x * nrm, __ATOMIC_RELAXED, __HIP_MEMORY_SCOPE_AGENT);
        __hip_atomic_fetch_add(a + 1, v.y * nrm, __ATOMIC_RELAXED, __HIP_MEMORY_SCOPE_AGENT);
        __hip_atomic_fetch_add(a + 2, v.z * nrm, __ATOMIC_RELAXED, __HIP_MEMORY_SCOPE_AGENT);
        __hip_atomic_fetch_add(a + 3, v.w * nrm, __ATOMIC_RELAXED, __HIP_MEMORY_SCOPE_AGENT);
    }
}

// ------------------------------------------------------------- WMMA GEMM
// OUT = act(A[M,128] @ Wt^T + bias), Wt pre-transposed f16 [128 cols][128 k].
// 256 threads = 8 waves; block owns a 16-row strip of A (staged in LDS as
// f16); wave w owns output columns [16w, 16w+16). K-loop: 4x
// v_wmma_f32_16x16x32_f16. Fragment K-pattern (e/8)*16 + half*8 + e%8 makes
// each half-fragment a contiguous 8xf16 run -> b128 loads for both A and B.
// DUAL: columns 0-63 -> (bias0,out0), 64-127 -> (bias1,out1), ld 64.
template <bool RELU, bool DUAL>
__global__ __launch_bounds__(256) void gemm_wmma_kernel(
    const float* __restrict__ A, const _Float16* __restrict__ Wt,
    const float* __restrict__ bias0, const float* __restrict__ bias1,
    float* __restrict__ out0, float* __restrict__ out1, int M) {
    __shared__ _Float16 As[16][LDSP];

    const int m0  = blockIdx.x * 16;
    const int tid = threadIdx.x;

    // stage the 16x128 A strip into LDS as f16: float4 loads, 2 iters/thread
#pragma unroll
    for (int it = 0; it < 2; ++it) {
        int i = tid + it * 256;          // 512 float4 chunks total
        int r  = i >> 5;                 // 32 chunks per row
        int k4 = (i & 31) << 2;
        int row = m0 + r; if (row >= M) row = M - 1;
        float4 v = *(const float4*)(A + (long long)row * DIN + k4);
        As[r][k4 + 0] = (_Float16)v.x;
        As[r][k4 + 1] = (_Float16)v.y;
        As[r][k4 + 2] = (_Float16)v.z;
        As[r][k4 + 3] = (_Float16)v.w;
    }
    __syncthreads();

    const int wave = tid >> 5;      // column tile 0..7
    const int lane = tid & 31;
    const int half = lane >> 4;     // K-half selector within a fragment
    const int mr   = lane & 15;     // A row / B-C column within tile
    const int colg = wave * 16 + mr;
    const _Float16* wrow = Wt + (long long)colg * DIN;

    v8f acc = {0.f, 0.f, 0.f, 0.f, 0.f, 0.f, 0.f, 0.f};

#pragma unroll
    for (int kc = 0; kc < DIN; kc += 32) {
        const int kb = kc + (half << 3);
        v8h a0 = *(const v8h*)&As[mr][kb];
        v8h a1 = *(const v8h*)&As[mr][kb + 16];
        v8h b0 = *(const v8h*)(wrow + kb);
        v8h b1 = *(const v8h*)(wrow + kb + 16);
        v16h a = __builtin_shufflevector(a0, a1, 0, 1, 2, 3, 4, 5, 6, 7,
                                                 8, 9, 10, 11, 12, 13, 14, 15);
        v16h b = __builtin_shufflevector(b0, b1, 0, 1, 2, 3, 4, 5, 6, 7,
                                                 8, 9, 10, 11, 12, 13, 14, 15);
        acc = __builtin_amdgcn_wmma_f32_16x16x32_f16(
            false, a, false, b, (short)0, acc, false, false);
    }

    const float bb = DUAL ? ((colg < DOUT) ? bias0[colg] : bias1[colg - DOUT])
                          : bias0[colg];
    // per-lane output base: row m0+8*half, column colg
    const int ldo = DUAL ? DOUT : DIN;
    float* obase;
    if (!DUAL)              obase = out0 + (long long)(m0 + 8 * half) * ldo + colg;
    else if (colg < DOUT)   obase = out0 + (long long)(m0 + 8 * half) * ldo + colg;
    else                    obase = out1 + (long long)(m0 + 8 * half) * ldo + (colg - DOUT);

    if (m0 + 16 <= M) {
        // uniform fast path: whole strip in bounds, no exec-mask churn
#pragma unroll
        for (int r = 0; r < 8; ++r) {
            float v = acc[r] + bb;
            if (RELU) v = fmaxf(v, 0.f);
            obase[(long long)r * ldo] = v;   // C/D layout: VGPR r -> M = r + 8*half
        }
    } else {
#pragma unroll
        for (int r = 0; r < 8; ++r) {
            if (m0 + 8 * half + r < M) {
                float v = acc[r] + bb;
                if (RELU) v = fmaxf(v, 0.f);
                obase[(long long)r * ldo] = v;
            }
        }
    }
}

// ---------------------------------------------------------------- launcher
extern "C" void kernel_launch(void* const* d_in, const int* in_sizes, int n_in,
                              void* d_out, int out_size, void* d_ws, size_t ws_size,
                              hipStream_t stream) {
    const float* x   = (const float*)d_in[0];
    const int*   ei  = (const int*)d_in[1];   // int32 per harness convention
    const float* W1  = (const float*)d_in[2];
    const float* b1  = (const float*)d_in[3];
    const float* Wmu = (const float*)d_in[4];
    const float* bmu = (const float*)d_in[5];
    const float* Wls = (const float*)d_in[6];
    const float* bls = (const float*)d_in[7];
    float* out = (float*)d_out;

    const int N = in_sizes[0] / DIN;   // 100000 nodes
    const int E = in_sizes[1] / 2;     // 1600000 edges
    const int* src = ei;
    const int* dst = ei + E;

    // ---- workspace layout (all 256B aligned) ----
    char* ws = (char*)d_ws;
    float* deg  = (float*)ws;                                          // N (becomes dinv)
    long long off = (((long long)N * 4) + 255) / 256 * 256;
    float* bufA = (float*)(ws + off);                                  // N*128 (aggX -> aggH)
    off += (long long)N * DIN * 4;
    float* bufH = (float*)(ws + off);                                  // N*128 (h)
    off += (long long)N * DIN * 4;
    _Float16* Wt1 = (_Float16*)(ws + off);                             // 128*128 f16
    off += (long long)DIN * DIN * 2;
    _Float16* Wt2 = (_Float16*)(ws + off);                             // 128*128 f16

    const long long feat_elems = (long long)N * DIN;
    const long long warps = (long long)(E + N);
    const int sblocks = (int)((warps * 32 + 255) / 256);

    // 1. init: deg=1 (self-loop), aggX=0, weights -> transposed f16
    fill_kernel<<<1024, 256, 0, stream>>>(deg, N, 1.0f);
    fill_kernel<<<4096, 256, 0, stream>>>(bufA, feat_elems, 0.0f);
    prep_w_kernel<<<(DIN * DIN + 255) / 256, 256, 0, stream>>>(W1, Wmu, Wls, Wt1, Wt2);
    // 2. degree accumulation over real edges
    degree_kernel<<<(E + 255) / 256, 256, 0, stream>>>(dst, E, deg);
    // 3. dinv = rsqrt(deg)
    rsqrt_kernel<<<(N + 255) / 256, 256, 0, stream>>>(deg, N);
    // 4. aggX = normalized scatter of x
    scatter128_kernel<<<sblocks, 256, 0, stream>>>(x, bufA, src, dst, deg, E, N);
    // 5. h = relu(aggX @ W1 + b1)
    gemm_wmma_kernel<true, false><<<(N + 15) / 16, 256, 0, stream>>>(
        bufA, Wt1, b1, nullptr, bufH, nullptr, N);
    // 6. reuse bufA as aggH: zero it
    fill_kernel<<<4096, 256, 0, stream>>>(bufA, feat_elems, 0.0f);
    // 7. aggH = normalized scatter of h
    scatter128_kernel<<<sblocks, 256, 0, stream>>>(bufH, bufA, src, dst, deg, E, N);
    // 8. [mu | logstd] = aggH @ [Wmu | Wls] + [bmu | bls] -> d_out directly
    gemm_wmma_kernel<false, true><<<(N + 15) / 16, 256, 0, stream>>>(
        bufA, Wt2, bmu, bls, out, out + (long long)N * DOUT, N);
}
